// CausalAttention_19524921328029
// MI455X (gfx1250) — matrix operand order repaired and checked by reading, not compiled, not run on hardware
//
#include <hip/hip_runtime.h>

#define DIMD 1024
#define SEQ  2048
#define NB   2
#define NH   16
#define HD   64
#define MTOT (NB * SEQ)   // 4096 token rows

typedef _Float16 v8h  __attribute__((ext_vector_type(8)));
typedef _Float16 v16h __attribute__((ext_vector_type(16)));
typedef float    v8f  __attribute__((ext_vector_type(8)));

__device__ __forceinline__ v8f vzero8f() {
  v8f v;
  for (int i = 0; i < 8; ++i) v[i] = 0.0f;
  return v;
}

// D = A x B + C, 16x16x32 f16 -> f32
__device__ __forceinline__ v8f wmma_f16(v16h a, v16h b, v8f c) {
  return __builtin_amdgcn_wmma_f32_16x16x32_f16(
      /*neg_a=*/false, a, /*neg_b=*/false, b,
      /*c_mod=*/(short)0, c, /*reuse_a=*/false, /*reuse_b=*/false);
}

// A-fragment (16x32 f16, M x K): lanes 0-15 hold rows M=0..15 with K=0..7,16..23;
// lanes 16-31 hold K=8..15,24..31. Row-major A, leading dim ld.
__device__ __forceinline__ v16h load_a16(const _Float16* __restrict__ base,
                                         int ld, int row0, int k0) {
  const int lane = threadIdx.x & 31;
  const _Float16* p = base + (size_t)(row0 + (lane & 15)) * ld + k0 + ((lane >> 4) << 3);
  v8h lo = *(const v8h*)p;
  v8h hi = *(const v8h*)(p + 16);
  v16h out;
  for (int i = 0; i < 8; ++i) { out[i] = lo[i]; out[i + 8] = hi[i]; }
  return out;
}

// B-fragment (32x16 f16, K x N) where B is stored "transposed": row n holds all K
// contiguously (i.e. memory layout [N, K] row-major, leading dim ld).
// lanes 0-15: col n, K=0..7,16..23 ; lanes 16-31: col n, K=8..15,24..31.
__device__ __forceinline__ v16h load_b16(const _Float16* __restrict__ base,
                                         int ld, int n0, int k0) {
  const int lane = threadIdx.x & 31;
  const _Float16* p = base + (size_t)(n0 + (lane & 15)) * ld + k0 + ((lane >> 4) << 3);
  v8h lo = *(const v8h*)p;
  v8h hi = *(const v8h*)(p + 16);
  v16h out;
  for (int i = 0; i < 8; ++i) { out[i] = lo[i]; out[i + 8] = hi[i]; }
  return out;
}

// ---------------- elementwise f32 -> f16 cast ----------------
__global__ void f32_to_f16_kernel(const float* __restrict__ in,
                                  _Float16* __restrict__ out, int n) {
  int i = blockIdx.x * blockDim.x + threadIdx.x;
  if (i < n) out[i] = (_Float16)in[i];
}

// ---------------- confounder bias: cb[b] = cw * mean(conf[b,:]) ----------------
__global__ void cbias_kernel(const float* __restrict__ conf,
                             const float* __restrict__ cw,
                             float* __restrict__ out) {
  int b = threadIdx.x;
  if (b < NB) {
    float s = 0.0f;
    for (int i = 0; i < 16; ++i) s += conf[b * 16 + i];
    out[b] = cw[0] * (s * (1.0f / 16.0f));
  }
}

// ---------------- WMMA GEMM: C[m,n] = sum_k A[m,k] * W[n,k] + bias[n] ----------
// Wave tile: 32x64 (2 A-fragments x 4 B-fragments = 8 WMMAs per k-step;
// each B-fragment reused twice -> ~21 flop/byte from L2).
// mode 0: store f16 into [B,H,S,Hd]    (Q, K)
// mode 1: store f16 into [B,H,Hd,S]    (V transposed for PV fragments)
// mode 2: store f32 into [M,N]         (final output projection)
__global__ __launch_bounds__(256) void gemm32_wmma(
    const _Float16* __restrict__ A, const _Float16* __restrict__ Bw,
    const float* __restrict__ bias, void* __restrict__ outp,
    int M, int N, int K, int mode) {
  const int wid = blockIdx.x * (blockDim.x >> 5) + (threadIdx.x >> 5);
  const int nSuper = N >> 6;
  const int m0 = (wid / nSuper) << 5;
  const int n0 = (wid % nSuper) << 6;
  if (m0 >= M) return;  // wave-uniform

  v8f acc[2][4];
  for (int r = 0; r < 2; ++r)
    for (int t = 0; t < 4; ++t) acc[r][t] = vzero8f();

  for (int k0 = 0; k0 < K; k0 += 32) {
    v16h a0 = load_a16(A, K, m0, k0);
    v16h a1 = load_a16(A, K, m0 + 16, k0);
#pragma unroll
    for (int t = 0; t < 4; ++t) {
      v16h b = load_b16(Bw, K, n0 + (t << 4), k0);
      acc[0][t] = wmma_f16(a0, b, acc[0][t]);
      acc[1][t] = wmma_f16(a1, b, acc[1][t]);
    }
  }

  const int lane = threadIdx.x & 31;
  const int half = lane >> 4, l15 = lane & 15;
#pragma unroll
  for (int r = 0; r < 2; ++r) {
#pragma unroll
    for (int t = 0; t < 4; ++t) {
      const int n = n0 + (t << 4) + l15;
      const float bn = bias[n];
#pragma unroll
      for (int i = 0; i < 8; ++i) {
        const int m = m0 + (r << 4) + i + (half << 3);
        const float val = acc[r][t][i] + bn;
        if (mode == 2) {
          ((float*)outp)[(size_t)m * N + n] = val;
        } else {
          const int bb = m >> 11, s = m & (SEQ - 1);
          const int h = n >> 6, hd = n & (HD - 1);
          _Float16* o16 = (_Float16*)outp;
          if (mode == 0)
            o16[(((size_t)(bb * NH + h) * SEQ) + s) * HD + hd] = (_Float16)val;
          else
            o16[(((size_t)(bb * NH + h) * HD) + hd) * SEQ + s] = (_Float16)val;
        }
      }
    }
  }
}

// ---------------- flash attention (one wave per (b,h,16-query tile)) -----------
__global__ __launch_bounds__(256) void flash_attn_wmma(
    const _Float16* __restrict__ q16, const _Float16* __restrict__ k16,
    const _Float16* __restrict__ vT16, const float* __restrict__ treat,
    const float* __restrict__ tw, const float* __restrict__ cbias,
    _Float16* __restrict__ attn16) {
  __shared__ _Float16 pst[8][16 * 32];  // per-wave P staging tile (16x32)

  const int wv = threadIdx.x >> 5;
  const int wid = blockIdx.x * (blockDim.x >> 5) + wv;
  const int qtilesPerHead = SEQ / 16;  // 128
  const int b = wid / (NH * qtilesPerHead);
  const int rest = wid % (NH * qtilesPerHead);
  const int h = rest / qtilesPerHead;
  const int q0 = (rest % qtilesPerHead) << 4;

  const int lane = threadIdx.x & 31, half = lane >> 4, l15 = lane & 15;
  const int bh = b * NH + h;
  const _Float16* qp = q16 + (size_t)bh * SEQ * HD;
  const _Float16* kp = k16 + (size_t)bh * SEQ * HD;
  const _Float16* vp = vT16 + (size_t)bh * HD * SEQ;
  const float cb = cbias[b];
  const float twv = tw[0];

  // preload Q A-fragments covering Hd=64 (two K-steps of 32)
  v16h qa0 = load_a16(qp, HD, q0, 0);
  v16h qa1 = load_a16(qp, HD, q0, 32);

  float mrow[8], lrow[8];
#pragma unroll
  for (int i = 0; i < 8; ++i) { mrow[i] = -1e30f; lrow[i] = 0.0f; }
  v8f o[4];
  for (int t = 0; t < 4; ++t) o[t] = vzero8f();

  _Float16* lp = pst[wv];

  for (int j0 = 0; j0 < SEQ; j0 += 32) {
    // ---- scores: S(16q x 32k) = Q(16x64) @ K_chunk^T, two 16-key subtiles
    v8f s0 = vzero8f(), s1 = vzero8f();
    v16h kb;
    kb = load_b16(kp, HD, j0, 0);        s0 = wmma_f16(qa0, kb, s0);
    kb = load_b16(kp, HD, j0, 32);       s0 = wmma_f16(qa1, kb, s0);
    kb = load_b16(kp, HD, j0 + 16, 0);   s1 = wmma_f16(qa0, kb, s1);
    kb = load_b16(kp, HD, j0 + 16, 32);  s1 = wmma_f16(qa1, kb, s1);

    if (j0 + 32 < SEQ) {
      __builtin_prefetch(kp + (size_t)(j0 + 32) * HD, 0, 1);
      __builtin_prefetch(vp + (size_t)(j0 + 32), 0, 1);
    }

    // per-key bias (treatment) + per-batch scalar (confounders)
    const float tb0 = twv * treat[b * SEQ + j0 + l15] + cb;
    const float tb1 = twv * treat[b * SEQ + j0 + 16 + l15] + cb;

    // ---- online softmax; row m = i + 8*half lives across the 16 lanes of a half
#pragma unroll
    for (int i = 0; i < 8; ++i) {
      float a0 = s0[i] * 0.125f + tb0;  // 1/sqrt(64)
      float a1 = s1[i] * 0.125f + tb1;
      float mx = fmaxf(a0, a1);
#pragma unroll
      for (int d = 1; d < 16; d <<= 1) mx = fmaxf(mx, __shfl_xor(mx, d, 32));
      const float mnew = fmaxf(mrow[i], mx);
      const float corr = __expf(mrow[i] - mnew);
      const float p0 = __expf(a0 - mnew);
      const float p1 = __expf(a1 - mnew);
      float rs = p0 + p1;
#pragma unroll
      for (int d = 1; d < 16; d <<= 1) rs += __shfl_xor(rs, d, 32);
      lrow[i] = lrow[i] * corr + rs;
      mrow[i] = mnew;
#pragma unroll
      for (int t = 0; t < 4; ++t) o[t][i] *= corr;
      // stage P (D-layout -> LDS, row-major 16x32)
      const int r = i + (half << 3);
      lp[r * 32 + l15] = (_Float16)p0;
      lp[r * 32 + 16 + l15] = (_Float16)p1;
    }
    asm volatile("" ::: "memory");  // keep ds stores before ds loads (in-order HW)

    // reload P as A-fragment (16x32)
    const _Float16* pap = lp + l15 * 32 + (half << 3);
    v8h plo = *(const v8h*)pap;
    v8h phi = *(const v8h*)(pap + 16);
    v16h pa;
    for (int i = 0; i < 8; ++i) { pa[i] = plo[i]; pa[i + 8] = phi[i]; }

    // ---- O += P(16x32) @ V_chunk(32x64): V transposed [Hd,S] gives contiguous frags
#pragma unroll
    for (int t = 0; t < 4; ++t) {
      v16h vb = load_b16(vp, SEQ, t << 4, j0);
      o[t] = wmma_f16(pa, vb, o[t]);
    }
  }

  // ---- epilogue: normalize and write merged-head f16 [B*S, D]
#pragma unroll
  for (int i = 0; i < 8; ++i) {
    const float inv = 1.0f / lrow[i];
    const int trow = b * SEQ + q0 + i + (half << 3);
#pragma unroll
    for (int t = 0; t < 4; ++t) {
      const int d = h * HD + (t << 4) + l15;
      attn16[(size_t)trow * DIMD + d] = (_Float16)(o[t][i] * inv);
    }
  }
}

extern "C" void kernel_launch(void* const* d_in, const int* in_sizes, int n_in,
                              void* d_out, int out_size, void* d_ws, size_t ws_size,
                              hipStream_t stream) {
  (void)in_sizes; (void)n_in; (void)out_size; (void)ws_size;
  const float* x  = (const float*)d_in[0];
  const float* tr = (const float*)d_in[1];
  const float* cf = (const float*)d_in[2];
  const float* Wq = (const float*)d_in[3];
  const float* bq = (const float*)d_in[4];
  const float* Wk = (const float*)d_in[5];
  const float* bk = (const float*)d_in[6];
  const float* Wv = (const float*)d_in[7];
  const float* bv = (const float*)d_in[8];
  const float* Wo = (const float*)d_in[9];
  const float* bo = (const float*)d_in[10];
  const float* tw = (const float*)d_in[11];
  const float* cw = (const float*)d_in[12];

  char* ws = (char*)d_ws;
  size_t off = 0;
  _Float16* xh   = (_Float16*)(ws + off); off += (size_t)MTOT * DIMD * 2;  // 8 MB
  _Float16* Wq16 = (_Float16*)(ws + off); off += (size_t)DIMD * DIMD * 2;  // 2 MB
  _Float16* Wk16 = (_Float16*)(ws + off); off += (size_t)DIMD * DIMD * 2;
  _Float16* Wv16 = (_Float16*)(ws + off); off += (size_t)DIMD * DIMD * 2;
  _Float16* Wo16 = (_Float16*)(ws + off); off += (size_t)DIMD * DIMD * 2;
  _Float16* q16  = (_Float16*)(ws + off); off += (size_t)MTOT * DIMD * 2;  // 8 MB
  _Float16* k16  = (_Float16*)(ws + off); off += (size_t)MTOT * DIMD * 2;
  _Float16* vT16 = (_Float16*)(ws + off); off += (size_t)MTOT * DIMD * 2;
  _Float16* at16 = (_Float16*)(ws + off); off += (size_t)MTOT * DIMD * 2;
  float* cbias   = (float*)(ws + off);    off += 2 * sizeof(float);

  const int nx = MTOT * DIMD;
  const int nw = DIMD * DIMD;
  f32_to_f16_kernel<<<(nx + 255) / 256, 256, 0, stream>>>(x, xh, nx);
  f32_to_f16_kernel<<<(nw + 255) / 256, 256, 0, stream>>>(Wq, Wq16, nw);
  f32_to_f16_kernel<<<(nw + 255) / 256, 256, 0, stream>>>(Wk, Wk16, nw);
  f32_to_f16_kernel<<<(nw + 255) / 256, 256, 0, stream>>>(Wv, Wv16, nw);
  f32_to_f16_kernel<<<(nw + 255) / 256, 256, 0, stream>>>(Wo, Wo16, nw);
  cbias_kernel<<<1, 32, 0, stream>>>(cf, cw, cbias);

  const int gemmBlocks = (MTOT / 32) * (DIMD / 64) / 8;  // 256 blocks of 8 waves
  gemm32_wmma<<<gemmBlocks, 256, 0, stream>>>(xh, Wq16, bq, q16, MTOT, DIMD, DIMD, 0);
  gemm32_wmma<<<gemmBlocks, 256, 0, stream>>>(xh, Wk16, bk, k16, MTOT, DIMD, DIMD, 0);
  gemm32_wmma<<<gemmBlocks, 256, 0, stream>>>(xh, Wv16, bv, vT16, MTOT, DIMD, DIMD, 1);

  const int attnBlocks = (NB * NH * (SEQ / 16)) / 8;     // 512
  flash_attn_wmma<<<attnBlocks, 256, 0, stream>>>(q16, k16, vT16, tr, tw, cbias, at16);

  gemm32_wmma<<<gemmBlocks, 256, 0, stream>>>(at16, Wo16, bo, d_out, MTOT, DIMD, DIMD, 2);
}
